// ConditionalVelocityModel_22840636080299
// MI455X (gfx1250) — matrix operand order repaired and checked by reading, not compiled, not run on hardware
//
#include <hip/hip_runtime.h>
#include <hip/hip_bf16.h>
#include <math.h>

#define B_   64
#define S_   1024
#define D_   256
#define COV_ 256
#define H_   128
#define E_   4
#define UW_  0.3f
#define N_   (E_ * D_)     /* 1024 */
#define ROWS (B_ * S_)     /* 65536 */

typedef __attribute__((ext_vector_type(16))) __bf16 v16bf;
typedef __attribute__((ext_vector_type(8)))  float  v8f;

__device__ __forceinline__ float gelu_exact(float x) {
    return 0.5f * x * (1.0f + erff(x * 0.70710678118654752440f));
}

// ---------------------------------------------------------------------------
// Kernel 1: per-batch covariate gating (importance, router, scale/shift nets).
// One block per batch element, 256 threads. Writes routing rows of d_out.
// ---------------------------------------------------------------------------
__global__ __launch_bounds__(256) void gating_kernel(
    const float* __restrict__ cov_emb,
    const float* __restrict__ wi1, const float* __restrict__ bi1,
    const float* __restrict__ wi2, const float* __restrict__ bi2,
    const float* __restrict__ ws1, const float* __restrict__ bs1,
    const float* __restrict__ ws2, const float* __restrict__ bs2,
    const float* __restrict__ wh1, const float* __restrict__ bh1,
    const float* __restrict__ wh2, const float* __restrict__ bh2,
    const float* __restrict__ wr1, const float* __restrict__ br1,
    const float* __restrict__ gr,  const float* __restrict__ br_ln,
    const float* __restrict__ wr2, const float* __restrict__ br2,
    const float* __restrict__ rt,
    float* __restrict__ ws_wspec, float* __restrict__ ws_wsum,
    float* __restrict__ ws_scale, float* __restrict__ ws_shift,
    float* __restrict__ out_routing)
{
    __shared__ float s_cov[COV_], s_t1[COV_], s_wc[COV_];
    __shared__ float s_h[H_], s_g[H_], s_s1[H_], s_h1[H_];
    __shared__ float s_stats[2];
    __shared__ float s_logit[E_ - 1];

    const int b = blockIdx.x;
    const int tid = threadIdx.x;

    s_cov[tid] = cov_emb[b * COV_ + tid];
    __syncthreads();

    // importance net layer 1 (tanh)
    float acc = bi1[tid];
    for (int i = 0; i < COV_; ++i) acc += s_cov[i] * wi1[tid * COV_ + i];
    s_t1[tid] = tanhf(acc);
    __syncthreads();

    // importance net layer 2 (sigmoid), weighted covariates
    acc = bi2[tid];
    for (int i = 0; i < COV_; ++i) acc += s_t1[i] * wi2[tid * COV_ + i];
    s_wc[tid] = s_cov[tid] * (1.0f / (1.0f + expf(-acc)));
    __syncthreads();

    // first layers of router / scale / shift nets (H outputs each)
    if (tid < H_) {
        float a1 = br1[tid], a2 = bs1[tid], a3 = bh1[tid];
        for (int i = 0; i < COV_; ++i) {
            float w = s_wc[i];
            a1 += w * wr1[tid * COV_ + i];
            a2 += w * ws1[tid * COV_ + i];
            a3 += w * wh1[tid * COV_ + i];
        }
        s_h[tid]  = a1;
        s_s1[tid] = fmaxf(a2, 0.0f);
        s_h1[tid] = fmaxf(a3, 0.0f);
    }
    __syncthreads();

    if (tid == 0) {
        float m = 0.0f;
        for (int i = 0; i < H_; ++i) m += s_h[i];
        m /= (float)H_;
        float v = 0.0f;
        for (int i = 0; i < H_; ++i) { float d = s_h[i] - m; v += d * d; }
        v /= (float)H_;
        s_stats[0] = m;
        s_stats[1] = rsqrtf(v + 1e-5f);
    }
    __syncthreads();

    if (tid < H_) {
        float y = (s_h[tid] - s_stats[0]) * s_stats[1] * gr[tid] + br_ln[tid];
        s_g[tid] = gelu_exact(y);
    }
    __syncthreads();

    if (tid < E_ - 1) {
        float a = br2[tid];
        for (int i = 0; i < H_; ++i) a += s_g[i] * wr2[tid * H_ + i];
        float rtc = fminf(fmaxf(rt[0], 0.1f), 5.0f);
        s_logit[tid] = a / rtc;
    }
    __syncthreads();

    if (tid == 0) {
        float mx = fmaxf(s_logit[0], fmaxf(s_logit[1], s_logit[2]));
        float e0 = expf(s_logit[0] - mx);
        float e1 = expf(s_logit[1] - mx);
        float e2 = expf(s_logit[2] - mx);
        float inv = (1.0f - UW_) / (e0 + e1 + e2);
        float w0 = e0 * inv, w1 = e1 * inv, w2 = e2 * inv;
        ws_wspec[b * 3 + 0] = w0;
        ws_wspec[b * 3 + 1] = w1;
        ws_wspec[b * 3 + 2] = w2;
        ws_wsum[b] = w0 + w1 + w2;
        out_routing[b * E_ + 0] = UW_;
        out_routing[b * E_ + 1] = w0;
        out_routing[b * E_ + 2] = w1;
        out_routing[b * E_ + 3] = w2;
    }

    // second layers of scale (sigmoid) / shift nets (D outputs)
    {
        float a2 = bs2[tid], a3 = bh2[tid];
        for (int i = 0; i < H_; ++i) {
            a2 += s_s1[i] * ws2[tid * H_ + i];
            a3 += s_h1[i] * wh2[tid * H_ + i];
        }
        ws_scale[b * D_ + tid] = 1.0f / (1.0f + expf(-a2));
        ws_shift[b * D_ + tid] = a3;
    }
}

// ---------------------------------------------------------------------------
// Kernel 2: load-balancing loss (tiny).
// ---------------------------------------------------------------------------
__global__ void loss_kernel(const float* __restrict__ ws_wspec,
                            float* __restrict__ out_loss)
{
    if (threadIdx.x == 0 && blockIdx.x == 0) {
        float loss = 0.0f;
        for (int e = 0; e < E_ - 1; ++e) {
            float u = 0.0f;
            for (int b = 0; b < B_; ++b) u += ws_wspec[b * 3 + e];
            u /= (float)B_;
            float d = u - 1.0f / (float)(E_ - 1);
            loss += d * d;
        }
        out_loss[0] = loss / (float)(E_ - 1);
    }
}

// ---------------------------------------------------------------------------
// Kernel 3: swizzle expert weights fp32 -> bf16 in WMMA B-fragment order.
// B[k][n] with n = e*D + o, k = i  ->  w_exp flat [n*256 + k].
// Fragment (tile nt, kstep kb): lane l holds column n = nt*16 + (l&15),
// K = kb*32 + ((l>>4)?8:0) + (j<8 ? j : j+8) for j = 0..15 (16 contiguous bf16 per lane).
// ---------------------------------------------------------------------------
__global__ __launch_bounds__(512) void wswz_kernel(const float* __restrict__ w_exp,
                                                   __bf16* __restrict__ wswz)
{
    const int blk  = blockIdx.x;        // nt*8 + kb
    const int nt   = blk >> 3;
    const int kb   = blk & 7;
    const int t    = threadIdx.x;
    const int lane = t & 31;
    const int j    = t >> 5;            // 0..15
    const int n    = nt * 16 + (lane & 15);
    const int k    = kb * 32 + ((lane >> 4) ? 8 : 0) + (j < 8 ? j : j + 8);
    wswz[((size_t)blk * 32 + lane) * 16 + j] = (__bf16)w_exp[(size_t)n * D_ + k];
}

// ---------------------------------------------------------------------------
// Kernel 4: fused expert GEMM (bf16 WMMA, fp32 accum) + LN + GELU + residual
// + time modulation + covariate mixing. One block = 16 rows x 1024 cols.
// 256 threads = 8 waves; each wave: 8 N-tiles x 8 K-steps of
// v_wmma_f32_16x16x32_bf16 with A fragments resident in VGPRs.
// ---------------------------------------------------------------------------
__global__ __launch_bounds__(256) void moe_gemm_kernel(
    const float*  __restrict__ x,
    const int*    __restrict__ t_idx,
    const float*  __restrict__ embed_table,
    const __bf16* __restrict__ wswz,
    const float*  __restrict__ b_exp,
    const float*  __restrict__ g_exp,
    const float*  __restrict__ be_exp,
    const float*  __restrict__ ws_wspec,
    const float*  __restrict__ ws_wsum,
    const float*  __restrict__ ws_scale,
    const float*  __restrict__ ws_shift,
    float* __restrict__ out)
{
    extern __shared__ char smem_raw[];
    __bf16* a_frag = (__bf16*)smem_raw;                 // 8*32*16 bf16 = 8 KB
    float*  hstage = (float*)(smem_raw + 8192);         // 16*1024 f32 = 64 KB
    float*  s_mu   = hstage + 16 * N_;                  // 64
    float*  s_rstd = s_mu + 64;                         // 64
    float*  s_misc = s_rstd + 64;                       // wspec[3], wsum

    const int tid  = threadIdx.x;
    const int lane = tid & 31;
    const int wave = tid >> 5;
    const int row0 = blockIdx.x * 16;
    const int b    = row0 >> 10;        // 16 | S, so one batch per block

    if (tid < 4) s_misc[tid] = (tid < 3) ? ws_wspec[b * 3 + tid] : ws_wsum[b];

    // ---- stage x tile into LDS in WMMA A-fragment order (fp32 -> bf16) ----
    {
        const int m  = tid & 15;
        const int kc = tid >> 4;        // 0..15 chunks of 16 K
        const float* xr = x + (size_t)(row0 + m) * D_ + kc * 16;
#pragma unroll
        for (int jj = 0; jj < 16; ++jj) {
            int k   = kc * 16 + jj;
            int kb  = k >> 5;
            int r   = k & 31;
            int grp = r >> 3;                       // 0..3
            int fl  = m + ((grp & 1) << 4);         // fragment lane
            int fj  = (r & 7) + ((grp >> 1) << 3);  // element in lane
            a_frag[((size_t)kb * 32 + fl) * 16 + fj] = (__bf16)xr[jj];
        }
    }
    __syncthreads();

    // ---- load all 8 A fragments once per wave (reused for all N-tiles) ----
    v16bf a[8];
#pragma unroll
    for (int kb = 0; kb < 8; ++kb)
        a[kb] = *(const v16bf*)&a_frag[((size_t)kb * 32 + lane) * 16];

    // ---- GEMM: 8 N-tiles per wave, full K = 256 ----
#pragma unroll 1
    for (int ntl = 0; ntl < 8; ++ntl) {
        const int nt   = wave * 8 + ntl;
        const int ncol = nt * 16 + (lane & 15);
        const float bias = b_exp[ncol];
        v8f c = { bias, bias, bias, bias, bias, bias, bias, bias };
        const __bf16* bp = wswz + (size_t)nt * 8 * 32 * 16 + (size_t)lane * 16;
        if (ntl < 7) __builtin_prefetch(bp + 8 * 32 * 16, 0, 0);
#pragma unroll
        for (int kb = 0; kb < 8; ++kb) {
            v16bf bfrag = *(const v16bf*)(bp + (size_t)kb * 32 * 16);
            c = __builtin_amdgcn_wmma_f32_16x16x32_bf16(
                    false, a[kb], false, bfrag, (short)0, c, false, false);
        }
        // C layout: element i -> row (lane<16 ? i : i+8), col = ncol
        const int rbase = (lane < 16) ? 0 : 8;
#pragma unroll
        for (int i = 0; i < 8; ++i)
            hstage[(size_t)(rbase + i) * N_ + ncol] = c[i];
    }
    __syncthreads();

    // ---- LayerNorm stats per (row, expert) ----
    if (tid < 64) {
        const int m = tid >> 2, e = tid & 3;
        const float* hp = hstage + (size_t)m * N_ + e * D_;
        float s = 0.0f, ss = 0.0f;
        for (int d = 0; d < D_; ++d) { float v = hp[d]; s += v; ss += v * v; }
        float mu  = s / (float)D_;
        float var = ss / (float)D_ - mu * mu;
        s_mu[tid]   = mu;
        s_rstd[tid] = rsqrtf(var + 1e-5f);
    }
    __syncthreads();

    // ---- fused epilogue: LN*g+b, GELU, +x, *gamma(t), expert mix ----
    {
        const int m  = tid >> 4;
        const int d0 = tid & 15;
        const int te = t_idx[b];
        const float sw0 = s_misc[0], sw1 = s_misc[1], sw2 = s_misc[2];
        const float swsum = s_misc[3];
        const float* hrow = hstage + (size_t)m * N_;
        const size_t rowoff = (size_t)(row0 + m) * D_;
#pragma unroll 1
        for (int ii = 0; ii < 16; ++ii) {
            const int d = d0 + ii * 16;
            const float xv  = x[rowoff + d];
            const float gam = embed_table[(size_t)te * D_ + d];
            float eo[4];
#pragma unroll
            for (int e = 0; e < 4; ++e) {
                float h = hrow[e * D_ + d];
                float y = (h - s_mu[m * 4 + e]) * s_rstd[m * 4 + e]
                          * g_exp[e * D_ + d] + be_exp[e * D_ + d];
                eo[e] = (gelu_exact(y) + xv) * gam;
            }
            float spec = sw0 * eo[1] + sw1 * eo[2] + sw2 * eo[3];
            out[rowoff + d] = UW_ * eo[0]
                            + ws_scale[b * D_ + d] * spec
                            + ws_shift[b * D_ + d] * swsum;
        }
    }
}

// ---------------------------------------------------------------------------
extern "C" void kernel_launch(void* const* d_in, const int* in_sizes, int n_in,
                              void* d_out, int out_size, void* d_ws, size_t ws_size,
                              hipStream_t stream)
{
    (void)in_sizes; (void)n_in; (void)out_size; (void)ws_size;

    const float* x      = (const float*)d_in[0];
    const int*   t_idx  = (const int*)  d_in[1];
    const float* cov    = (const float*)d_in[2];
    const float* etab   = (const float*)d_in[3];
    const float* w_exp  = (const float*)d_in[4];
    const float* b_exp  = (const float*)d_in[5];
    const float* g_exp  = (const float*)d_in[6];
    const float* be_exp = (const float*)d_in[7];
    const float* wi1 = (const float*)d_in[8],  *bi1   = (const float*)d_in[9];
    const float* wi2 = (const float*)d_in[10], *bi2   = (const float*)d_in[11];
    const float* ws1 = (const float*)d_in[12], *bs1   = (const float*)d_in[13];
    const float* ws2 = (const float*)d_in[14], *bs2   = (const float*)d_in[15];
    const float* wh1 = (const float*)d_in[16], *bh1   = (const float*)d_in[17];
    const float* wh2 = (const float*)d_in[18], *bh2   = (const float*)d_in[19];
    const float* wr1 = (const float*)d_in[20], *br1   = (const float*)d_in[21];
    const float* gr  = (const float*)d_in[22], *br_ln = (const float*)d_in[23];
    const float* wr2 = (const float*)d_in[24], *br2   = (const float*)d_in[25];
    const float* rt  = (const float*)d_in[26];

    float* out         = (float*)d_out;              // [B*S*D]
    float* out_loss    = out + (size_t)B_ * S_ * D_; // [1]
    float* out_routing = out_loss + 1;               // [B*E]

    char* ws = (char*)d_ws;
    __bf16* wswz    = (__bf16*)ws;  ws += (size_t)N_ * D_ * sizeof(__bf16); // 512 KB
    float* ws_wspec = (float*)ws;   ws += (size_t)B_ * 3 * sizeof(float);
    float* ws_wsum  = (float*)ws;   ws += (size_t)B_ * sizeof(float);
    float* ws_scale = (float*)ws;   ws += (size_t)B_ * D_ * sizeof(float);
    float* ws_shift = (float*)ws;   ws += (size_t)B_ * D_ * sizeof(float);

    gating_kernel<<<B_, 256, 0, stream>>>(
        cov, wi1, bi1, wi2, bi2, ws1, bs1, ws2, bs2,
        wh1, bh1, wh2, bh2, wr1, br1, gr, br_ln, wr2, br2, rt,
        ws_wspec, ws_wsum, ws_scale, ws_shift, out_routing);

    loss_kernel<<<1, 64, 0, stream>>>(ws_wspec, out_loss);

    wswz_kernel<<<(N_ / 16) * 8, 512, 0, stream>>>(w_exp, wswz);

    const size_t shbytes = 8192 + (size_t)16 * N_ * 4 + 64 * 4 + 64 * 4 + 16;
    moe_gemm_kernel<<<ROWS / 16, 256, shbytes, stream>>>(
        x, t_idx, etab, wswz, b_exp, g_exp, be_exp,
        ws_wspec, ws_wsum, ws_scale, ws_shift, out);
}
